// DrBC_8083128451161
// MI455X (gfx1250) — matrix operand order repaired and checked by reading, not compiled, not run on hardware
//
#include <hip/hip_runtime.h>
#include <hip/hip_bf16.h>

#define NN 50000
#define EE 800000
#define DD 128

typedef __attribute__((ext_vector_type(16))) _Float16 v16h;
typedef __attribute__((ext_vector_type(8)))  _Float16 v8h;
typedef __attribute__((ext_vector_type(8)))  float    v8f;

#define WMMA16(a, b, c) __builtin_amdgcn_wmma_f32_16x16x32_f16( \
    false, (a), false, (b), (short)0, (c), false, false)

// ---- fragment loaders ------------------------------------------------------
// A (16x32 f16, M x K): lane holds row M=lane&15; halves 0..15 = K (lane/16)*16+0..15.
// arow = A + row*128.
__device__ __forceinline__ v16h load_a_frag(const _Float16* __restrict__ arow,
                                            int k0, int lane) {
  const _Float16* p = arow + k0 + ((lane >> 4) << 4);
  v8h lo = *(const v8h*)(p);
  v8h hi = *(const v8h*)(p + 8);
  v16h a;
#pragma unroll
  for (int i = 0; i < 8; ++i) { a[i] = lo[i]; a[i + 8] = hi[i]; }
  return a;
}

// B (32x16 f16, K x N): lane holds column N=lane&15; 16 contiguous halves at
// K = k0 + (lane/16)*16.  wrow = W + n*128 (W is [ncols,128] row-major, K inner).
__device__ __forceinline__ v16h load_b_frag(const _Float16* __restrict__ wrow,
                                            int k0, int lane) {
  return *(const v16h*)(wrow + k0 + ((lane >> 4) << 4));
}

__device__ __forceinline__ float sigmoid_f(float x) {
  return 1.0f / (1.0f + expf(-x));
}

// ---- small elementwise kernels --------------------------------------------
__global__ void k_fc1(const float* __restrict__ x, const float* __restrict__ w,
                      const float* __restrict__ b, float* __restrict__ h,
                      _Float16* __restrict__ h16) {
  int i = blockIdx.x * blockDim.x + threadIdx.x;   // N*128 threads exact
  int n = i >> 7, d = i & 127;
  float acc = b[d] + x[n * 3 + 0] * w[d * 3 + 0]
                   + x[n * 3 + 1] * w[d * 3 + 1]
                   + x[n * 3 + 2] * w[d * 3 + 2];
  acc = acc > 0.0f ? acc : 0.01f * acc;
  h[i] = acc;
  h16[i] = (_Float16)acc;
}

__global__ void k_fill1(float* __restrict__ p, int count) {
  int i = blockIdx.x * blockDim.x + threadIdx.x;
  if (i < count) p[i] = 1.0f;
}

__global__ void k_deg_count(const int* __restrict__ row, float* __restrict__ deg) {
  int e = blockIdx.x * blockDim.x + threadIdx.x;
  if (e < EE) unsafeAtomicAdd(&deg[row[e]], 1.0f);
}

__global__ void k_rsqrt(const float* __restrict__ deg, float* __restrict__ dinv,
                        int count) {
  int i = blockIdx.x * blockDim.x + threadIdx.x;
  if (i < count) dinv[i] = rsqrtf(deg[i]);
}

__global__ void k_cvt_f16(const float* __restrict__ in, _Float16* __restrict__ out,
                          int count) {
  int i = blockIdx.x * blockDim.x + threadIdx.x;
  if (i < count) out[i] = (_Float16)in[i];
}

// agg[i] = dinv[n]^2 * hW[i]   (self-loop contribution doubles as the zero-init)
__global__ void k_agg_init(const float* __restrict__ dinv,
                           const float* __restrict__ hW,
                           float* __restrict__ agg) {
  int i = blockIdx.x * blockDim.x + threadIdx.x;   // N*128 exact
  float s = dinv[i >> 7];
  agg[i] = s * s * hW[i];
}

// one wave per edge: gather hW[row], scale, atomic-scatter into agg[col]
__global__ void k_scatter(const int* __restrict__ row, const int* __restrict__ col,
                          const float* __restrict__ dinv,
                          const float* __restrict__ hW, float* __restrict__ agg) {
  int e = blockIdx.x * (blockDim.x >> 5) + (threadIdx.x >> 5);
  if (e >= EE) return;
  int lane = threadIdx.x & 31;
  int r = row[e], c = col[e];
  float nrm = dinv[r] * dinv[c];
  const float4 v = *(const float4*)(hW + r * DD + lane * 4);
  float* dst = agg + c * DD + lane * 4;
  unsafeAtomicAdd(dst + 0, nrm * v.x);
  unsafeAtomicAdd(dst + 1, nrm * v.y);
  unsafeAtomicAdd(dst + 2, nrm * v.z);
  unsafeAtomicAdd(dst + 3, nrm * v.w);
}

__global__ void k_fc3(const float* __restrict__ y, const float* __restrict__ w,
                      const float* __restrict__ b, float* __restrict__ out) {
  int n = blockIdx.x * blockDim.x + threadIdx.x;
  if (n >= NN) return;
  float acc = b[0];
#pragma unroll 8
  for (int d = 0; d < 64; ++d) acc += y[n * 64 + d] * w[d];
  out[n] = acc;
}

// ---- WMMA GEMM: C[M,ncols] = A[M,128] @ W[ncols,128]^T + bias (opt leaky) ---
// grid.x = M/16 row tiles; wave tw handles column tile tw (blockDim = 32*ntiles)
__global__ void k_gemm_f16(const _Float16* __restrict__ A,
                           const _Float16* __restrict__ W,
                           const float* __restrict__ bias,
                           float* __restrict__ C, int ldc, int leaky) {
  int lane = threadIdx.x & 31;
  int wv   = threadIdx.x >> 5;
  int m0   = blockIdx.x * 16;
  int n0   = wv * 16;
  int coln = n0 + (lane & 15);
  const _Float16* arow = A + (m0 + (lane & 15)) * 128;
  const _Float16* wrow = W + coln * 128;
  float bv = bias[coln];
  v8f c;
#pragma unroll
  for (int i = 0; i < 8; ++i) c[i] = bv;
#pragma unroll
  for (int kt = 0; kt < 4; ++kt) {
    v16h a = load_a_frag(arow, kt * 32, lane);
    v16h b = load_b_frag(wrow, kt * 32, lane);
    c = WMMA16(a, b, c);
  }
  int rbase = m0 + ((lane >> 4) << 3);
#pragma unroll
  for (int v = 0; v < 8; ++v) {
    float val = c[v];
    if (leaky) val = val > 0.0f ? val : 0.01f * val;
    C[(rbase + v) * ldc + coln] = val;
  }
}

// ---- fused GRU step: 6 GEMM tiles + gates, in-register ---------------------
// grid.x = N/16; 8 waves/block, wave wv owns 16 columns n0=wv*16 of D=128.
__global__ void k_gru(const _Float16* __restrict__ X16,
                      const _Float16* __restrict__ H16,
                      const _Float16* __restrict__ Wih,   // [384,128] f16
                      const _Float16* __restrict__ Whh,   // [384,128] f16
                      const float* __restrict__ bih, const float* __restrict__ bhh,
                      float* __restrict__ H, _Float16* __restrict__ H16out,
                      float* __restrict__ Hmax, int first) {
  int lane = threadIdx.x & 31;
  int wv   = threadIdx.x >> 5;
  int m0   = blockIdx.x * 16;
  int coln = wv * 16 + (lane & 15);
  const _Float16* xrow = X16 + (m0 + (lane & 15)) * 128;
  const _Float16* hrow = H16 + (m0 + (lane & 15)) * 128;
  const _Float16* wir = Wih + coln * 128;
  const _Float16* wiz = Wih + (128 + coln) * 128;
  const _Float16* win = Wih + (256 + coln) * 128;
  const _Float16* whr = Whh + coln * 128;
  const _Float16* whz = Whh + (128 + coln) * 128;
  const _Float16* whn = Whh + (256 + coln) * 128;

  v8f cir = {}, ciz = {}, cin = {}, chr_ = {}, chz = {}, chn = {};
#pragma unroll
  for (int kt = 0; kt < 4; ++kt) {
    int k0 = kt * 32;
    v16h ax = load_a_frag(xrow, k0, lane);
    v16h ah = load_a_frag(hrow, k0, lane);
    cir  = WMMA16(ax, load_b_frag(wir, k0, lane), cir);
    ciz  = WMMA16(ax, load_b_frag(wiz, k0, lane), ciz);
    cin  = WMMA16(ax, load_b_frag(win, k0, lane), cin);
    chr_ = WMMA16(ah, load_b_frag(whr, k0, lane), chr_);
    chz  = WMMA16(ah, load_b_frag(whz, k0, lane), chz);
    chn  = WMMA16(ah, load_b_frag(whn, k0, lane), chn);
  }
  float bir = bih[coln], biz = bih[128 + coln], bin = bih[256 + coln];
  float bhr = bhh[coln], bhz = bhh[128 + coln], bhn = bhh[256 + coln];
  int rbase = m0 + ((lane >> 4) << 3);
#pragma unroll
  for (int v = 0; v < 8; ++v) {
    int idx = (rbase + v) * DD + coln;
    float r  = sigmoid_f(cir[v] + bir + chr_[v] + bhr);
    float z  = sigmoid_f(ciz[v] + biz + chz[v] + bhz);
    float nn = tanhf(cin[v] + bin + r * (chn[v] + bhn));
    float hnew = (1.0f - z) * nn + z * H[idx];
    H[idx] = hnew;
    H16out[idx] = (_Float16)hnew;
    Hmax[idx] = first ? hnew : fmaxf(Hmax[idx], hnew);
  }
}

// ---- host-side orchestration ----------------------------------------------
extern "C" void kernel_launch(void* const* d_in, const int* in_sizes, int n_in,
                              void* d_out, int out_size, void* d_ws, size_t ws_size,
                              hipStream_t stream) {
  const float* x       = (const float*)d_in[0];
  const int*   edge    = (const int*)d_in[1];
  const float* fc1_w   = (const float*)d_in[2];
  const float* fc1_b   = (const float*)d_in[3];
  const float* gcn_w   = (const float*)d_in[4];
  const float* gcn_b   = (const float*)d_in[5];
  const float* gru_wih = (const float*)d_in[6];
  const float* gru_whh = (const float*)d_in[7];
  const float* gru_bih = (const float*)d_in[8];
  const float* gru_bhh = (const float*)d_in[9];
  const float* fc2_w   = (const float*)d_in[10];
  const float* fc2_b   = (const float*)d_in[11];
  const float* fc3_w   = (const float*)d_in[12];
  const float* fc3_b   = (const float*)d_in[13];
  float* out = (float*)d_out;

  const int* row = edge;
  const int* col = edge + EE;

  // workspace carve (256-B aligned)
  char* p = (char*)d_ws;
  auto carve = [&](size_t bytes) {
    char* r = p;
    p += (bytes + 255) & ~(size_t)255;
    return r;
  };
  float* h    = (float*)carve((size_t)NN * DD * 4);
  float* hmax = (float*)carve((size_t)NN * DD * 4);
  float* hW   = (float*)carve((size_t)NN * DD * 4);
  float* agg  = (float*)carve((size_t)NN * DD * 4);
  float* y    = (float*)carve((size_t)NN * 64 * 4);
  float* deg  = (float*)carve((size_t)NN * 4);
  float* dinv = (float*)carve((size_t)NN * 4);
  _Float16* h16a   = (_Float16*)carve((size_t)NN * DD * 2);
  _Float16* h16b   = (_Float16*)carve((size_t)NN * DD * 2);
  _Float16* x16    = (_Float16*)carve((size_t)NN * DD * 2);
  _Float16* hmax16 = (_Float16*)carve((size_t)NN * DD * 2);
  _Float16* wg16   = (_Float16*)carve((size_t)5 * 128 * 128 * 2);
  _Float16* wih16  = (_Float16*)carve((size_t)5 * 384 * 128 * 2);
  _Float16* whh16  = (_Float16*)carve((size_t)5 * 384 * 128 * 2);
  _Float16* fc2w16 = (_Float16*)carve((size_t)64 * 128 * 2);

  const int ND = NN * DD;               // 6,400,000
  const int BLK = 256;
  const int G_ND = ND / BLK;            // 25000 (exact)

  // one-time weight conversion to f16
  k_cvt_f16<<<(5 * 128 * 128 + BLK - 1) / BLK, BLK, 0, stream>>>(gcn_w, wg16, 5 * 128 * 128);
  k_cvt_f16<<<(5 * 384 * 128 + BLK - 1) / BLK, BLK, 0, stream>>>(gru_wih, wih16, 5 * 384 * 128);
  k_cvt_f16<<<(5 * 384 * 128 + BLK - 1) / BLK, BLK, 0, stream>>>(gru_whh, whh16, 5 * 384 * 128);
  k_cvt_f16<<<(64 * 128 + BLK - 1) / BLK, BLK, 0, stream>>>(fc2_w, fc2w16, 64 * 128);

  // fc1 + leaky -> h, h16a
  k_fc1<<<G_ND, BLK, 0, stream>>>(x, fc1_w, fc1_b, h, h16a);

  // degrees (self-loops => init 1) and dinv
  k_fill1<<<(NN + BLK - 1) / BLK, BLK, 0, stream>>>(deg, NN);
  k_deg_count<<<(EE + BLK - 1) / BLK, BLK, 0, stream>>>(row, deg);
  k_rsqrt<<<(NN + BLK - 1) / BLK, BLK, 0, stream>>>(deg, dinv, NN);

  const int RT = NN / 16;               // 3125 row tiles
  for (int l = 0; l < 5; ++l) {
    _Float16* cur = (l & 1) ? h16b : h16a;
    _Float16* nxt = (l & 1) ? h16a : h16b;
    // GCN linear: hW = h @ W^T + b   (8 col-tile waves per block)
    k_gemm_f16<<<RT, 256, 0, stream>>>(cur, wg16 + (size_t)l * 128 * 128,
                                       gcn_b + l * 128, hW, DD, 0);
    // self-loop init + edge scatter (L2-resident atomics)
    k_agg_init<<<G_ND, BLK, 0, stream>>>(dinv, hW, agg);
    k_scatter<<<EE / 8, 256, 0, stream>>>(row, col, dinv, hW, agg);
    k_cvt_f16<<<G_ND, BLK, 0, stream>>>(agg, x16, ND);
    // fused GRU + running max
    k_gru<<<RT, 256, 0, stream>>>(x16, cur, wih16 + (size_t)l * 384 * 128,
                                  whh16 + (size_t)l * 384 * 128,
                                  gru_bih + l * 384, gru_bhh + l * 384,
                                  h, nxt, hmax, l == 0);
  }

  // head: fc2 (WMMA, leaky) then fc3 dot
  k_cvt_f16<<<G_ND, BLK, 0, stream>>>(hmax, hmax16, ND);
  k_gemm_f16<<<RT, 128, 0, stream>>>(hmax16, fc2w16, fc2_b, y, 64, 1);
  k_fc3<<<(NN + BLK - 1) / BLK, BLK, 0, stream>>>(y, fc3_w, fc3_b, out);
}
